// WindowAttention_9543417332052
// MI455X (gfx1250) — matrix operand order repaired and checked by reading, not compiled, not run on hardware
//
#include <hip/hip_runtime.h>

typedef __attribute__((ext_vector_type(16))) _Float16 v16h;
typedef __attribute__((ext_vector_type(8)))  float    v8f;
typedef __attribute__((ext_vector_type(4)))  unsigned int u32x4;

#define DIM    128
#define NH     4
#define N_TOK  49
#define HD     32
#define NPAD   64
#define NWIN   64
#define SCALE_F 0.17677669529663687f   // 32^-0.5

// LDS strides in elements (all chosen so every WMMA operand read is 16B-aligned)
#define XH_STR 136
#define QK_STR 264
#define VT_STR 72
#define AF_STR 68
#define AH_STR 72
#define OH_STR 136

#define XH_BYTES (64 * XH_STR * 2)            // 17408
#define QK_BYTES (64 * QK_STR * 2)            // 33792
#define VT_BYTES (NH * HD * VT_STR * 2)       // 18432
#define AF_BYTES (NH * NPAD * AF_STR * 4)     // 69632
#define AH_BYTES (NH * NPAD * AH_STR * 2)     // 36864
#define OH_BYTES (64 * OH_STR * 2)            // 17408
#define SMEM_BYTES (XH_BYTES + QK_BYTES + VT_BYTES + AF_BYTES + AH_BYTES + OH_BYTES) // 193536

union V16u { u32x4 u[2]; v16h h; };
union Pk4  { _Float16 h[4]; uint2 u; };

// A-operand (16x32 f16): lanes 0-15 -> M=lane, K={k0+0..7, k0+16..23};
// lanes 16-31 -> M=lane-16, K={k0+8..15, k0+24..31}. Two 16B loads per lane.
__device__ __forceinline__ v16h ldA(const _Float16* base, int stride, int mtile, int k0, int lane) {
    int m  = mtile + (lane & 15);
    int hi = lane >> 4;
    const _Float16* p = base + m * stride + k0 + hi * 8;
    V16u r;
    r.u[0] = *(const u32x4*)p;           // K +0..7
    r.u[1] = *(const u32x4*)(p + 16);    // K +16..23
    return r.h;
}

// B-operand (32x16 f16): lanes 0-15 -> N=lane, K={k0+0..7, k0+16..23};
// lanes 16-31 -> N=lane-16, K={k0+8..15, k0+24..31}. Operand source laid out [N][K].
__device__ __forceinline__ v16h ldB(const _Float16* base, int stride, int ntile, int k0, int lane) {
    int n  = ntile + (lane & 15);
    int hi = lane >> 4;
    const _Float16* p = base + n * stride + k0 + hi * 8;
    V16u r;
    r.u[0] = *(const u32x4*)p;           // K +0..7
    r.u[1] = *(const u32x4*)(p + 16);    // K +16..23
    return r.h;
}

__device__ __forceinline__ v8f wmma_f16(v16h a, v16h b, v8f c) {
    return __builtin_amdgcn_wmma_f32_16x16x32_f16(false, a, false, b, (short)0, c, false, false);
}

extern "C" __global__ void wa_convert_weights(const float* __restrict__ qkv_w,
                                              const float* __restrict__ proj_w,
                                              _Float16* __restrict__ wsH) {
    int idx = blockIdx.x * 256 + threadIdx.x;
    const int NQ = 3 * DIM * DIM;         // 49152
    if (idx < NQ) {
        wsH[idx] = (_Float16)qkv_w[idx];
    } else if (idx < NQ + DIM * DIM) {
        wsH[idx] = (_Float16)proj_w[idx - NQ];
    }
}

extern "C" __global__ __launch_bounds__(128, 1)
void wa_fused_kernel(const float* __restrict__ x,
                     const float* __restrict__ mask,
                     const float* __restrict__ qkv_b,
                     const float* __restrict__ proj_b,
                     const float* __restrict__ rpb_table,
                     const int*   __restrict__ rel_index,
                     const _Float16* __restrict__ wqkv,   // [384][128] f16
                     const _Float16* __restrict__ wproj,  // [128][128] f16
                     float* __restrict__ out) {
    extern __shared__ char smem[];
    _Float16* xh  = (_Float16*)(smem);
    _Float16* qkh = (_Float16*)(smem + XH_BYTES);
    _Float16* vT  = (_Float16*)(smem + XH_BYTES + QK_BYTES);
    float*    aF  = (float*)   (smem + XH_BYTES + QK_BYTES + VT_BYTES);
    _Float16* aH  = (_Float16*)(smem + XH_BYTES + QK_BYTES + VT_BYTES + AF_BYTES);
    _Float16* oh  = (_Float16*)(smem + XH_BYTES + QK_BYTES + VT_BYTES + AF_BYTES + AH_BYTES);

    const int b    = blockIdx.x;
    const int tid  = threadIdx.x;
    const int lane = tid & 31;
    const int wave = tid >> 5;
    const int hi   = lane >> 4;
    const float* xb = x + (size_t)b * (N_TOK * DIM);

    // ---- Stage 0: x -> LDS f16 (float4 loads, 4xf16 packed stores); zero pad rows
    {
        const float4* xb4 = (const float4*)xb;
        for (int idx = tid; idx < (N_TOK * DIM) / 4; idx += 128) {
            int r = idx >> 5, c = (idx & 31) * 4;
            float4 f = xb4[idx];
            Pk4 pk;
            pk.h[0] = (_Float16)f.x; pk.h[1] = (_Float16)f.y;
            pk.h[2] = (_Float16)f.z; pk.h[3] = (_Float16)f.w;
            *(uint2*)&xh[r * XH_STR + c] = pk.u;
        }
        Pk4 z; z.h[0] = z.h[1] = z.h[2] = z.h[3] = (_Float16)0.0f;
        for (int idx = tid; idx < ((NPAD - N_TOK) * DIM) / 4; idx += 128) {
            int r = N_TOK + (idx >> 5), c = (idx & 31) * 4;
            *(uint2*)&xh[r * XH_STR + c] = z.u;
        }
    }
    __syncthreads();

    // ---- Stage 1: QKV = x @ Wqkv^T + b
    // Tile split: nt = wave + i*4; i 0-1 -> q tiles (nt 0..7), i 2-3 -> k (8..15),
    // i 4-5 -> v (16..23). Uniform epilogue per loop => no EXEC divergence.

    // q tiles: pre-scale by hd^-0.5, store row-major
    for (int i = 0; i < 2; ++i) {
        int nt = wave + i * 4;
        v16h Bt[4];
        #pragma unroll
        for (int ks = 0; ks < 4; ++ks) Bt[ks] = ldB(wqkv, DIM, nt * 16, ks * 32, lane);
        int colg = nt * 16 + (lane & 15);
        float bias = qkv_b[colg];
        for (int mt = 0; mt < 4; ++mt) {
            v8f acc = {};
            #pragma unroll
            for (int ks = 0; ks < 4; ++ks)
                acc = wmma_f16(ldA(xh, XH_STR, mt * 16, ks * 32, lane), Bt[ks], acc);
            #pragma unroll
            for (int r = 0; r < 8; ++r) {
                int row = mt * 16 + r + hi * 8;
                qkh[row * QK_STR + colg] = (_Float16)((acc[r] + bias) * SCALE_F);
            }
        }
    }
    // k tiles: store row-major at column offset DIM
    for (int i = 2; i < 4; ++i) {
        int nt = wave + i * 4;
        v16h Bt[4];
        #pragma unroll
        for (int ks = 0; ks < 4; ++ks) Bt[ks] = ldB(wqkv, DIM, nt * 16, ks * 32, lane);
        int colg = nt * 16 + (lane & 15);            // 128..255
        float bias = qkv_b[colg];
        for (int mt = 0; mt < 4; ++mt) {
            v8f acc = {};
            #pragma unroll
            for (int ks = 0; ks < 4; ++ks)
                acc = wmma_f16(ldA(xh, XH_STR, mt * 16, ks * 32, lane), Bt[ks], acc);
            #pragma unroll
            for (int r = 0; r < 8; ++r) {
                int row = mt * 16 + r + hi * 8;
                qkh[row * QK_STR + colg] = (_Float16)(acc[r] + bias);
            }
        }
    }
    // v tiles: store transposed per head
    for (int i = 4; i < 6; ++i) {
        int nt = wave + i * 4;
        v16h Bt[4];
        #pragma unroll
        for (int ks = 0; ks < 4; ++ks) Bt[ks] = ldB(wqkv, DIM, nt * 16, ks * 32, lane);
        int colg = nt * 16 + (lane & 15);            // 256..383
        float bias = qkv_b[colg];
        int cc = colg - 2 * DIM;
        int h = cc >> 5, d = cc & 31;
        _Float16* vcol = vT + (h * HD + d) * VT_STR;
        for (int mt = 0; mt < 4; ++mt) {
            v8f acc = {};
            #pragma unroll
            for (int ks = 0; ks < 4; ++ks)
                acc = wmma_f16(ldA(xh, XH_STR, mt * 16, ks * 32, lane), Bt[ks], acc);
            #pragma unroll
            for (int r = 0; r < 8; ++r) {
                int row = mt * 16 + r + hi * 8;
                vcol[row] = (_Float16)(acc[r] + bias);
            }
        }
    }
    __syncthreads();

    // ---- Stage 2: scores = q @ k^T (head = wave)
    {
        int h = wave;
        const _Float16* qbase = qkh + h * HD;
        const _Float16* kbase = qkh + DIM + h * HD;
        for (int mt = 0; mt < 4; ++mt) {
            v16h A = ldA(qbase, QK_STR, mt * 16, 0, lane);
            for (int nt = 0; nt < 4; ++nt) {
                v16h B = ldB(kbase, QK_STR, nt * 16, 0, lane);
                v8f acc = {};
                acc = wmma_f16(A, B, acc);
                int col = nt * 16 + (lane & 15);
                #pragma unroll
                for (int r = 0; r < 8; ++r) {
                    int row = mt * 16 + r + hi * 8;
                    aF[(h * NPAD + row) * AF_STR + col] = acc[r];
                }
            }
        }
    }
    __syncthreads();

    // ---- Stage 3: softmax with fused rel-pos bias + window mask; emit f16
    {
        int h = wave;
        const float* maskb = mask + (size_t)(b & (NWIN - 1)) * (N_TOK * N_TOK);
        for (int rr = 0; rr < 2; ++rr) {
            int i = lane + rr * 32;
            if (i < N_TOK) {
                float* rowp = aF + (h * NPAD + i) * AF_STR;
                const int*   relrow = rel_index + i * N_TOK;
                const float* mrow   = maskb + i * N_TOK;
                float mx = -3.0e38f;
                for (int j = 0; j < N_TOK; ++j) {
                    float v = rowp[j] + rpb_table[relrow[j] * NH + h] + mrow[j];
                    rowp[j] = v;
                    mx = fmaxf(mx, v);
                }
                float s = 0.0f;
                for (int j = 0; j < N_TOK; ++j) {
                    float e = __expf(rowp[j] - mx);
                    rowp[j] = e;
                    s += e;
                }
                float inv = 1.0f / s;
                _Float16* ahrow = aH + (h * NPAD + i) * AH_STR;
                for (int j = 0; j < N_TOK; ++j) ahrow[j] = (_Float16)(rowp[j] * inv);
                for (int j = N_TOK; j < NPAD; ++j) ahrow[j] = (_Float16)0.0f;
            } else if (i < NPAD) {
                _Float16* ahrow = aH + (h * NPAD + i) * AH_STR;
                for (int j = 0; j < NPAD; ++j) ahrow[j] = (_Float16)0.0f;
            }
        }
    }
    __syncthreads();

    // ---- Stage 4: out_h = attn @ v  (head = wave)
    {
        int h = wave;
        const _Float16* abase = aH + h * NPAD * AH_STR;
        const _Float16* vbase = vT + h * HD * VT_STR;
        for (int mt = 0; mt < 4; ++mt) {
            for (int dt = 0; dt < 2; ++dt) {
                v8f acc = {};
                #pragma unroll
                for (int ks = 0; ks < 2; ++ks) {
                    v16h A = ldA(abase, AH_STR, mt * 16, ks * 32, lane);
                    v16h B = ldB(vbase, VT_STR, dt * 16, ks * 32, lane);
                    acc = wmma_f16(A, B, acc);
                }
                int col = dt * 16 + (lane & 15);
                #pragma unroll
                for (int r = 0; r < 8; ++r) {
                    int row = mt * 16 + r + hi * 8;
                    oh[row * OH_STR + h * HD + col] = (_Float16)acc[r];
                }
            }
        }
    }
    __syncthreads();

    // ---- Stage 5: projection + bias, f32 store
    float* outb = out + (size_t)b * (N_TOK * DIM);
    for (int i = 0; i < 2; ++i) {
        int nt = wave + i * 4;                    // 8 N-tiles over 4 waves
        v16h Bt[4];
        #pragma unroll
        for (int ks = 0; ks < 4; ++ks) Bt[ks] = ldB(wproj, DIM, nt * 16, ks * 32, lane);

        int col = nt * 16 + (lane & 15);
        float bias = proj_b[col];
        for (int mt = 0; mt < 4; ++mt) {
            v8f acc = {};
            #pragma unroll
            for (int ks = 0; ks < 4; ++ks)
                acc = wmma_f16(ldA(oh, OH_STR, mt * 16, ks * 32, lane), Bt[ks], acc);
            #pragma unroll
            for (int r = 0; r < 8; ++r) {
                int row = mt * 16 + r + hi * 8;
                if (row < N_TOK) outb[row * DIM + col] = acc[r] + bias;
            }
        }
    }
}

extern "C" void kernel_launch(void* const* d_in, const int* in_sizes, int n_in,
                              void* d_out, int out_size, void* d_ws, size_t ws_size,
                              hipStream_t stream) {
    const float* x         = (const float*)d_in[0];
    const float* mask      = (const float*)d_in[1];
    const float* qkv_w     = (const float*)d_in[2];
    const float* qkv_b     = (const float*)d_in[3];
    const float* proj_w    = (const float*)d_in[4];
    const float* proj_b    = (const float*)d_in[5];
    const float* rpb_table = (const float*)d_in[6];
    const int*   rel_index = (const int*)d_in[7];
    float* out = (float*)d_out;

    _Float16* wsH   = (_Float16*)d_ws;            // 384*128 + 128*128 halves = 128 KB
    _Float16* wqkv  = wsH;
    _Float16* wproj = wsH + 3 * DIM * DIM;

    // Opt-in to >64KB dynamic LDS (WGP has 320KB); deterministic, not a stream op.
    (void)hipFuncSetAttribute((const void*)wa_fused_kernel,
                              hipFuncAttributeMaxDynamicSharedMemorySize, SMEM_BYTES);

    // Weight f32->f16 conversion into workspace (deterministic every call).
    wa_convert_weights<<<(3 * DIM * DIM + DIM * DIM + 255) / 256, 256, 0, stream>>>(qkv_w, proj_w, wsH);

    // One workgroup (4 wave32) per window.
    wa_fused_kernel<<<8192, 128, SMEM_BYTES, stream>>>(
        x, mask, qkv_b, proj_b, rpb_table, rel_index, wqkv, wproj, out);
}
